// GVPGNNEncoder_36275293782057
// MI455X (gfx1250) — matrix operand order repaired and checked by reading, not compiled, not run on hardware
//
#include <hip/hip_runtime.h>
#include <math.h>

// ============================================================================
// GVP-GNN encoder for MI455X (gfx1250, wave32).
//
// Heavy dense ops run through one WMMA GEMM kernel (v_wmma_f32_16x16x32_f16,
// f16 in / f32 accumulate).  All GEMM dims are PADDED (K to x32, N to x64,
// M already x64) so the kernel has zero bounds checks: LDS tile fills are
// pure 16-byte global_load_b128 -> ds_store_b128, fragments are ds_load_b128,
// matching the CDNA5 WMMA VGPR layouts.  Edge rows (30720) processed in 8
// chunks of 3840 so chunk scratch (~110 MB) + f16 weights (~33 MB) stay
// resident in the 192 MB L2.
//
// Assumptions (compile-only): d_in flattened in dict-insertion order
// (74 tensors); ws_size >= ~220 MB.
// ============================================================================

constexpr int Bb   = 2;
constexpr int Nn   = 512;
constexpr int NBn  = Bb * Nn;          // 1024 nodes
constexpr int Ssz  = 1024;             // S
constexpr int Vsz  = 256;              // V
constexpr int ESz  = 32;               // ES
constexpr int Kk   = 30;               // K neighbors
constexpr int NLay = 3;
constexpr int EMB  = 512;
constexpr int LAT  = 256;
constexpr int NE   = NBn * Kk;         // 30720 edges
constexpr int HM   = 2 * Vsz + 1;      // 513 hidden vector channels (msg)
constexpr int SCAT = 2 * Ssz + ESz;    // 2080 scalar concat width
constexpr int NCH  = 8;                // edge chunks
constexpr int MCn  = (NBn / NCH) * Kk; // 3840 edge rows per chunk
constexpr int NBV  = NBn * Vsz;
// padded GEMM dims
constexpr int KNE  = 288;              // node-embed Ws K (262 -> 288)
constexpr int KMS  = 2624;             // msg Ws K (2593 -> 2624)
constexpr int KHM  = 544;              // msg Wh/Wv K (513 -> 544)
constexpr int NHM  = 576;              // msg Wh N (513 -> 576)

typedef __attribute__((ext_vector_type(16))) _Float16 v16h;
typedef __attribute__((ext_vector_type(8)))  _Float16 v8h;
typedef __attribute__((ext_vector_type(8)))  float    v8f;

// ----------------------------- device helpers ------------------------------
__device__ __forceinline__ float san(float x) {          // jnp.nan_to_num
  if (__builtin_isnan(x)) return 0.0f;
  if (x >  3.0e38f) return  3.0e38f;
  if (x < -3.0e38f) return -3.0e38f;
  return x;
}
__device__ __forceinline__ void ld3(const float* p, float* v) {
  v[0] = san(p[0]); v[1] = san(p[1]); v[2] = san(p[2]);
}
__device__ __forceinline__ float dot3(const float* a, const float* b) {
  return a[0]*b[0] + a[1]*b[1] + a[2]*b[2];
}
__device__ __forceinline__ void cross3(const float* a, const float* b, float* o) {
  o[0] = a[1]*b[2] - a[2]*b[1];
  o[1] = a[2]*b[0] - a[0]*b[2];
  o[2] = a[0]*b[1] - a[1]*b[0];
}
__device__ __forceinline__ void nrm3(const float* a, float* o) {   // x/sqrt(|x|^2+1e-8)
  float r = 1.0f / sqrtf(dot3(a, a) + 1e-8f);
  o[0] = a[0]*r; o[1] = a[1]*r; o[2] = a[2]*r;
}
__device__ __forceinline__ float sigm(float x) { return 1.0f / (1.0f + expf(-x)); }

// ============================================================================
// Unguarded WMMA GEMM on padded operands:
//   C[M x N](f32, ld=ldc) = act( A[M x K](f16, ld=lda) * Bt[N x K](f16) + bias )
//   Grid = (N/64, M/64); K, lda multiples of 32/8; everything in-bounds by
//   construction (zero padding), so tile fills are pure 16B vector loads.
// Block = 128 threads = 4 waves; macro-tile 64x64; 4 wmma per wave per k-step.
// Fragment layouts as loaded here (same k order for A and B):
//   A lane L: halves 0..7 <- K=16*hi+0..7, halves 8..15 <- K=16*hi+8..15
//   B lane L: halves e    <- K=16*hi+e,   column = L%16
//   C vgpr r: row = r + 8*hi, col = lane%16
// ============================================================================
__global__ __launch_bounds__(128)
void gemm_wmma(const _Float16* __restrict__ A, const _Float16* __restrict__ Bt,
               const float* __restrict__ bias, float* __restrict__ C,
               _Float16* __restrict__ Ch,
               int Kd, int lda, int ldc, int act)
{
  __shared__ __align__(16) _Float16 As[64][32];
  __shared__ __align__(16) _Float16 Bs[64][32];
  const int tid  = threadIdx.x;
  const int lane = tid & 31;
  const int wv   = tid >> 5;
  const int hi   = lane >> 4;
  const int lm   = lane & 15;
  const int m0   = blockIdx.y * 64;
  const int n0   = blockIdx.x * 64;

  v8f acc0 = {}, acc1 = {}, acc2 = {}, acc3 = {};

  for (int k0 = 0; k0 < Kd; k0 += 32) {
    // 64x32 halves per tile = 256 16-byte chunks; 128 threads -> 2 chunks each
#pragma unroll
    for (int t = tid; t < 256; t += 128) {
      int r = t >> 2, c8 = (t & 3) << 3;
      *(v8h*)&As[r][c8] = *(const v8h*)&A[(size_t)(m0 + r) * lda + k0 + c8];
      *(v8h*)&Bs[r][c8] = *(const v8h*)&Bt[(size_t)(n0 + r) * Kd + k0 + c8];
    }
    if (k0 + 32 < Kd) {                       // hint next k-slice toward L2
      int r = tid & 63;
      __builtin_prefetch(&Bt[(size_t)(n0 + r) * Kd + k0 + 32], 0, 1);
      __builtin_prefetch(&A[(size_t)(m0 + r) * lda + k0 + 32], 0, 1);
    }
    __syncthreads();

    v16h af;
    const int row = wv * 16 + lm;
    ((v8h*)&af)[0] = *(const v8h*)&As[row][16 * hi];
    ((v8h*)&af)[1] = *(const v8h*)&As[row][16 * hi + 8];
    v16h b0 = *(const v16h*)&Bs[ 0 + lm][16 * hi];
    v16h b1 = *(const v16h*)&Bs[16 + lm][16 * hi];
    v16h b2 = *(const v16h*)&Bs[32 + lm][16 * hi];
    v16h b3 = *(const v16h*)&Bs[48 + lm][16 * hi];

    acc0 = __builtin_amdgcn_wmma_f32_16x16x32_f16(false, af, false, b0, (short)0, acc0, false, false);
    acc1 = __builtin_amdgcn_wmma_f32_16x16x32_f16(false, af, false, b1, (short)0, acc1, false, false);
    acc2 = __builtin_amdgcn_wmma_f32_16x16x32_f16(false, af, false, b2, (short)0, acc2, false, false);
    acc3 = __builtin_amdgcn_wmma_f32_16x16x32_f16(false, af, false, b3, (short)0, acc3, false, false);
    __syncthreads();
  }

  const int gmBase = m0 + wv * 16 + 8 * hi;
#pragma unroll
  for (int j = 0; j < 4; ++j) {
    v8f a = (j == 0) ? acc0 : (j == 1) ? acc1 : (j == 2) ? acc2 : acc3;
    int gn = n0 + j * 16 + lm;
    float bv = bias ? bias[gn] : 0.0f;
#pragma unroll
    for (int r = 0; r < 8; ++r) {
      int gm = gmBase + r;
      float v = a[r] + bv;
      if (act == 1)      v = fmaxf(v, 0.0f);
      else if (act == 2) v = sigm(v);
      C[(size_t)gm * ldc + gn] = v;
      if (Ch) Ch[(size_t)gm * ldc + gn] = (_Float16)v;
    }
  }
}

// --------------------------- utility kernels -------------------------------
__global__ void k_f2h(const float* __restrict__ x, _Float16* __restrict__ y, int n) {
  int i = blockIdx.x * blockDim.x + threadIdx.x;
  if (i < n) y[i] = (_Float16)x[i];
}
// W (K x N) f32 row-major -> Wt (Np x Kp) f16, zero-padded
__global__ void k_wT(const float* __restrict__ W, _Float16* __restrict__ Wt,
                     int K, int N, int Kp, int Np) {
  int i = blockIdx.x * blockDim.x + threadIdx.x;
  if (i >= Np * Kp) return;
  int n = i / Kp, k = i % Kp;
  Wt[i] = (k < K && n < N) ? (_Float16)W[(size_t)k * N + n] : (_Float16)0.0f;
}

// --------------------------- geometry kernel -------------------------------
__global__ void k_geom(const float* __restrict__ coords, const float* __restrict__ seq_mask,
                       const float* __restrict__ coord_mask,
                       float* __restrict__ node_s, float* __restrict__ node_v,
                       float* __restrict__ nf)
{
  int n = blockIdx.x * blockDim.x + threadIdx.x;
  if (n >= NBn) return;
  int bI = n / Nn, nl = n % Nn;
  const float* cb = coords + (size_t)bI * Nn * 9;      // Xf[t] = cb + t*3

  // dihedral features (pad front 1, back 2 -> zeros -> cos=1,sin=0)
  for (int j = 0; j < 3; ++j) {
    int a = 3 * nl + j;
    float D = 0.0f;
    if (a >= 1 && a <= 3 * Nn - 3) {
      int t = a - 1;
      float X0[3], X1[3], X2[3], X3[3];
      ld3(cb + (size_t)(t    ) * 3, X0);
      ld3(cb + (size_t)(t + 1) * 3, X1);
      ld3(cb + (size_t)(t + 2) * 3, X2);
      ld3(cb + (size_t)(t + 3) * 3, X3);
      float d0[3] = {X1[0]-X0[0], X1[1]-X0[1], X1[2]-X0[2]};
      float d1[3] = {X2[0]-X1[0], X2[1]-X1[1], X2[2]-X1[2]};
      float d2[3] = {X3[0]-X2[0], X3[1]-X2[1], X3[2]-X2[2]};
      float u2[3], u1[3], u0[3]; nrm3(d0, u2); nrm3(d1, u1); nrm3(d2, u0);
      float c2[3], c1[3]; cross3(u2, u1, c2); cross3(u1, u0, c1);
      float n2[3], n1[3]; nrm3(c2, n2); nrm3(c1, n1);
      float cosD = fminf(fmaxf(dot3(n2, n1), -1.0f + 1e-7f), 1.0f - 1e-7f);
      float sg = dot3(u2, n1);
      float sgn = (sg > 0.f) ? 1.f : ((sg < 0.f) ? -1.f : 0.f);
      D = sgn * acosf(cosD);
    }
    node_s[n * 6 + j]     = cosf(D);
    node_s[n * 6 + 3 + j] = sinf(D);
  }

  float Natm[3], CAc[3], Cc[3];
  ld3(cb + (size_t)(nl * 3 + 0) * 3, Natm);
  ld3(cb + (size_t)(nl * 3 + 1) * 3, CAc);
  ld3(cb + (size_t)(nl * 3 + 2) * 3, Cc);

  float fwd[3] = {0, 0, 0}, bwd[3] = {0, 0, 0};
  if (nl < Nn - 1) {
    float nx[3]; ld3(cb + (size_t)((nl + 1) * 3 + 1) * 3, nx);
    float d[3] = {nx[0]-CAc[0], nx[1]-CAc[1], nx[2]-CAc[2]}; nrm3(d, fwd);
  }
  if (nl > 0) {
    float px[3]; ld3(cb + (size_t)((nl - 1) * 3 + 1) * 3, px);
    float d[3] = {px[0]-CAc[0], px[1]-CAc[1], px[2]-CAc[2]}; nrm3(d, bwd);
  }
  float t1[3] = {Natm[0]-CAc[0], Natm[1]-CAc[1], Natm[2]-CAc[2]};
  float t2[3] = {Cc[0]-CAc[0],   Cc[1]-CAc[1],   Cc[2]-CAc[2]};
  float nv[3], cv[3]; nrm3(t1, nv); nrm3(t2, cv);
  float sm3[3] = {nv[0]+cv[0], nv[1]+cv[1], nv[2]+cv[2]};
  float bis[3]; nrm3(sm3, bis);
  float cx[3]; cross3(nv, cv, cx);
  float perp[3]; nrm3(cx, perp);
  const float r3 = 0.57735026919f, r23 = 0.81649658092f;
  for (int c = 0; c < 3; ++c) {
    node_v[n * 9 + 0 * 3 + c] = fwd[c];
    node_v[n * 9 + 1 * 3 + c] = bwd[c];
    node_v[n * 9 + 2 * 3 + c] = -bis[c] * r3 - perp[c] * r23;
  }
  bool ok = seq_mask[n] > 0.5f;
  for (int c = 0; c < 3; ++c) ok = ok && (coord_mask[n * 3 + c] > 0.5f);
  nf[n] = ok ? 1.0f : 0.0f;
}

// ------------------------------ brute KNN ----------------------------------
__global__ void k_knn(const float* __restrict__ coords, const float* __restrict__ nf,
                      int* __restrict__ knn_idx)
{
  int n = blockIdx.x * blockDim.x + threadIdx.x;
  if (n >= NBn) return;
  int bI = n / Nn, nl = n % Nn;
  const float* cb = coords + (size_t)bI * Nn * 9;
  float ci[3]; ld3(cb + (size_t)(nl * 3 + 1) * 3, ci);
  bool oki = nf[n] > 0.5f;
  float bd[Kk]; int bidx[Kk];
  for (int k = 0; k < Kk; ++k) { bd[k] = 1e30f; bidx[k] = 0; }
  for (int j = 0; j < Nn; ++j) {
    float cj[3]; ld3(cb + (size_t)(j * 3 + 1) * 3, cj);
    float dx = ci[0]-cj[0], dy = ci[1]-cj[1], dz = ci[2]-cj[2];
    float d2 = dx*dx + dy*dy + dz*dz;
    if (!(oki && nf[bI * Nn + j] > 0.5f)) d2 = 1e10f;
    if (d2 < bd[Kk - 1]) {                       // stable insert (ties keep lower j)
      int p = Kk - 1;
      while (p > 0 && bd[p - 1] > d2) { bd[p] = bd[p - 1]; bidx[p] = bidx[p - 1]; --p; }
      bd[p] = d2; bidx[p] = j;
    }
  }
  for (int k = 0; k < Kk; ++k) knn_idx[n * Kk + k] = bidx[k];
}

// ------------------- edge features + tiny edge-GVP (h=1) -------------------
__global__ void k_edge(const float* __restrict__ coords, const float* __restrict__ nf,
                       const int* __restrict__ knn_idx,
                       const float* __restrict__ eeWh, const float* __restrict__ eeWs,
                       const float* __restrict__ eeBs, const float* __restrict__ eeWv,
                       const float* __restrict__ eeWg, const float* __restrict__ eeBg,
                       _Float16* __restrict__ edge_s_h, float* __restrict__ edge_v,
                       float* __restrict__ w_e)
{
  int e = blockIdx.x * blockDim.x + threadIdx.x;
  if (e >= NE) return;
  int n = e / Kk, bI = n / Nn, nl = n % Nn;
  int jl = knn_idx[e];
  int nj = bI * Nn + jl;
  const float* cb = coords + (size_t)bI * Nn * 9;
  float ci[3], cj[3];
  ld3(cb + (size_t)(nl * 3 + 1) * 3, ci);
  ld3(cb + (size_t)(jl * 3 + 1) * 3, cj);
  float dv[3] = {cj[0]-ci[0], cj[1]-ci[1], cj[2]-ci[2]};
  float d2 = dv[0]*dv[0] + dv[1]*dv[1] + dv[2]*dv[2];
  float dist = sqrtf(d2 + 1e-8f);
  float inv = 1.0f / dist;
  float dn[3] = {dv[0]*inv, dv[1]*inv, dv[2]*inv};

  float rbf[16];
  for (int m = 0; m < 16; ++m) {
    float z = (dist - (20.0f / 15.0f) * (float)m) / 1.25f;
    rbf[m] = expf(-z * z);
  }
  float wh0 = eeWh[0];
  float vh[3] = {dn[0]*wh0, dn[1]*wh0, dn[2]*wh0};
  float vn = sqrtf(vh[0]*vh[0] + vh[1]*vh[1] + vh[2]*vh[2] + 1e-8f);

  float gacc = eeBg[0];
  float soA[ESz];
  for (int o = 0; o < ESz; ++o) {
    float so = eeBs[o];
    for (int i2 = 0; i2 < 16; ++i2) so += rbf[i2] * eeWs[i2 * ESz + o];
    so += vn * eeWs[16 * ESz + o];
    soA[o] = so;
    gacc += so * eeWg[o];
  }
  float gate = sigm(gacc);
  float wv0 = eeWv[0];
  for (int o = 0; o < ESz; ++o) edge_s_h[(size_t)e * ESz + o] = (_Float16)fmaxf(soA[o], 0.0f);
  for (int c = 0; c < 3; ++c)   edge_v[(size_t)e * 3 + c] = vh[c] * wv0 * gate;
  w_e[e] = nf[n] * nf[nj];
}

__global__ void k_deg(const float* __restrict__ w_e, float* __restrict__ deg) {
  int n = blockIdx.x * blockDim.x + threadIdx.x;
  if (n >= NBn) return;
  float sm = 0.0f;
  for (int k = 0; k < Kk; ++k) sm += w_e[n * Kk + k];
  deg[n] = fmaxf(sm, 1.0f);
}

// ---------------- node-embed: Vh (K=3 einsum) + scat build -----------------
__global__ void k_ne_pre(const float* __restrict__ node_s, const float* __restrict__ node_v,
                         const float* __restrict__ neWh,
                         float* __restrict__ neVh_f, _Float16* __restrict__ neVh_h,
                         _Float16* __restrict__ ne_scat)
{
  int i = blockIdx.x * blockDim.x + threadIdx.x;
  if (i >= NBn * 256) return;
  int n = i >> 8, h = i & 255;
  float vh[3];
  for (int c = 0; c < 3; ++c) {
    float a = 0.0f;
    for (int i3 = 0; i3 < 3; ++i3) a += node_v[n * 9 + i3 * 3 + c] * neWh[i3 * 256 + h];
    vh[c] = a;
    neVh_f[(size_t)c * NBn * 256 + n * 256 + h] = a;
    neVh_h[(size_t)c * NBn * 256 + n * 256 + h] = (_Float16)a;
  }
  float vn = sqrtf(vh[0]*vh[0] + vh[1]*vh[1] + vh[2]*vh[2] + 1e-8f);
  ne_scat[(size_t)n * KNE + 6 + h] = (_Float16)vn;
  if (h < 6)  ne_scat[(size_t)n * KNE + h] = (_Float16)node_s[n * 6 + h];
  if (h < KNE - 262) ne_scat[(size_t)n * KNE + 262 + h] = (_Float16)0.0f;  // K pad
}

__global__ void k_s_init(const float* __restrict__ so, const float* __restrict__ nf,
                         float* __restrict__ s) {
  int i = blockIdx.x * blockDim.x + threadIdx.x;
  if (i < NBn * Ssz) s[i] = fmaxf(so[i], 0.0f) * nf[i >> 10];
}
__global__ void k_v_init(const float* __restrict__ Vo, const float* __restrict__ gate,
                         const float* __restrict__ nf, float* __restrict__ Vv) {
  int i = blockIdx.x * blockDim.x + threadIdx.x;
  if (i >= 3 * NBV) return;
  int rest = i % NBV;
  Vv[i] = Vo[i] * gate[rest] * nf[rest >> 8];
}

// ------------------- message-GVP input assembly (per chunk) ----------------
__global__ void k_msg_scat(const _Float16* __restrict__ s_h, const _Float16* __restrict__ edge_s_h,
                           const int* __restrict__ knn_idx, _Float16* __restrict__ scat, int n0)
{
  int col = blockIdx.x * blockDim.x + threadIdx.x;
  int r = blockIdx.y;
  if (col >= SCAT) return;
  int e = n0 * Kk + r;
  int n = n0 + r / Kk;
  _Float16 v;
  if (col < Ssz)              v = s_h[(size_t)n * Ssz + col];
  else if (col < Ssz + ESz)   v = edge_s_h[(size_t)e * ESz + (col - Ssz)];
  else {
    int nj = (n / Nn) * Nn + knn_idx[e];
    v = s_h[(size_t)nj * Ssz + (col - Ssz - ESz)];
  }
  scat[(size_t)r * KMS + col] = v;
}

__global__ void k_msg_vcat(const float* __restrict__ Vv, const float* __restrict__ edge_v,
                           const int* __restrict__ knn_idx, _Float16* __restrict__ Vcat, int n0)
{
  int col = blockIdx.x * blockDim.x + threadIdx.x;
  int r = blockIdx.y, c = blockIdx.z;
  if (col >= KHM) return;            // cols 513..543 are K padding -> 0
  int e = n0 * Kk + r;
  int n = n0 + r / Kk;
  float v = 0.0f;
  if (col < Vsz)        v = Vv[(size_t)c * NBV + n * Vsz + col];
  else if (col == Vsz)  v = edge_v[(size_t)e * 3 + c];
  else if (col < HM) {
    int nj = (n / Nn) * Nn + knn_idx[e];
    v = Vv[(size_t)c * NBV + nj * Vsz + (col - Vsz - 1)];
  }
  Vcat[(size_t)c * MCn * KHM + (size_t)r * KHM + col] = (_Float16)v;
}

// vn columns of the msg scat matrix (+ zero K-padding cols 2593..2623)
__global__ void k_msg_vn(const float* __restrict__ Vh_f, _Float16* __restrict__ scat)
{
  int col = blockIdx.x * blockDim.x + threadIdx.x;
  int r = blockIdx.y;
  if (col >= KMS - SCAT) return;     // 544 columns
  float v = 0.0f;
  if (col < HM) {
    float a = Vh_f[(size_t)0 * MCn * NHM + (size_t)r * NHM + col];
    float b = Vh_f[(size_t)1 * MCn * NHM + (size_t)r * NHM + col];
    float c = Vh_f[(size_t)2 * MCn * NHM + (size_t)r * NHM + col];
    v = sqrtf(a*a + b*b + c*c + 1e-8f);
  }
  scat[(size_t)r * KMS + SCAT + col] = (_Float16)v;
}

// --------------------------- aggregation (per chunk) -----------------------
__global__ void k_agg_s(const float* __restrict__ mso, const float* __restrict__ w_e,
                        const float* __restrict__ deg, float* __restrict__ agg_s, int n0)
{
  int col = blockIdx.x * blockDim.x + threadIdx.x;
  int row = blockIdx.y;
  if (col >= Ssz) return;
  int node = n0 + row;
  float acc = 0.0f;
  for (int k = 0; k < Kk; ++k) {
    int r = row * Kk + k;
    acc += fmaxf(mso[(size_t)r * Ssz + col], 0.0f) * w_e[n0 * Kk + r];
  }
  agg_s[(size_t)node * Ssz + col] = acc / deg[node];
}

__global__ void k_agg_v(const float* __restrict__ mVo, const float* __restrict__ mgate,
                        const float* __restrict__ w_e, const float* __restrict__ deg,
                        float* __restrict__ agg_v, int n0)
{
  int col = blockIdx.x * blockDim.x + threadIdx.x;
  int row = blockIdx.y, c = blockIdx.z;
  if (col >= Vsz) return;
  int node = n0 + row;
  float acc = 0.0f;
  for (int k = 0; k < Kk; ++k) {
    int r = row * Kk + k;
    acc += mVo[(size_t)c * MCn * Vsz + (size_t)r * Vsz + col]
         * mgate[(size_t)r * Vsz + col] * w_e[n0 * Kk + r];
  }
  agg_v[(size_t)c * NBV + node * Vsz + col] = acc / deg[node];
}

__global__ void k_addv(float* __restrict__ Vv, const float* __restrict__ agg_v) {
  int i = blockIdx.x * blockDim.x + threadIdx.x;
  if (i < 3 * NBV) Vv[i] += agg_v[i];
}

// -------------------------------- LayerNorm --------------------------------
// x[n] <- LN(x[n] + (reluAdd? relu(add):add), g, b) * rowscale[n]; D in {512,1024}
__global__ __launch_bounds__(256)
void k_ln(float* __restrict__ x, const float* __restrict__ add,
          const float* __restrict__ g, const float* __restrict__ bparam,
          const float* __restrict__ rowscale, int D, int reluAdd)
{
  int n = blockIdx.x, t = threadIdx.x;
  __shared__ float s1[256], s2[256];
  float yv[4];
  int cnt = D / 256;
  float a = 0.0f, b2 = 0.0f;
  for (int q = 0; q < cnt; ++q) {
    int d = t + q * 256;
    float y = x[(size_t)n * D + d];
    if (add) {
      float av = add[(size_t)n * D + d];
      if (reluAdd) av = fmaxf(av, 0.0f);
      y += av;
    }
    yv[q] = y; a += y; b2 += y * y;
  }
  s1[t] = a; s2[t] = b2; __syncthreads();
  for (int o = 128; o > 0; o >>= 1) {
    if (t < o) { s1[t] += s1[t + o]; s2[t] += s2[t + o]; }
    __syncthreads();
  }
  float mu = s1[0] / (float)D;
  float var = s2[0] / (float)D - mu * mu;
  float inv = rsqrtf(var + 1e-5f);
  float sc = rowscale ? rowscale[n] : 1.0f;
  for (int q = 0; q < cnt; ++q) {
    int d = t + q * 256;
    x[(size_t)n * D + d] = ((yv[q] - mu) * inv * g[d] + bparam[d]) * sc;
  }
}

// ---------------------------- feed-forward bits ----------------------------
__global__ void k_ff_scat(const _Float16* __restrict__ s_h, const float* __restrict__ fVh,
                          _Float16* __restrict__ fscat)
{
  int i = blockIdx.x * blockDim.x + threadIdx.x;
  if (i >= NBn * 1280) return;
  int n = i / 1280, col = i % 1280;
  if (col < Ssz) fscat[i] = s_h[(size_t)n * Ssz + col];
  else {
    int h = col - Ssz;
    float a = fVh[(size_t)0 * NBV + n * Vsz + h];
    float b = fVh[(size_t)1 * NBV + n * Vsz + h];
    float c = fVh[(size_t)2 * NBV + n * Vsz + h];
    fscat[i] = (_Float16)sqrtf(a*a + b*b + c*c + 1e-8f);
  }
}

__global__ void k_v_ff(float* __restrict__ Vv, const float* __restrict__ fVo,
                       const float* __restrict__ fgate, const float* __restrict__ nf) {
  int i = blockIdx.x * blockDim.x + threadIdx.x;
  if (i >= 3 * NBV) return;
  int rest = i % NBV;
  Vv[i] = (Vv[i] + fVo[i] * fgate[rest]) * nf[rest >> 8];
}

// --------------------------------- head ------------------------------------
__global__ void k_pool(const float* __restrict__ x, const float* __restrict__ seq_mask,
                       float* __restrict__ pooled)
{
  int i = blockIdx.x * blockDim.x + threadIdx.x;
  if (i >= Bb * EMB) return;
  int b = i / EMB, col = i % EMB;
  float smsum = 0.0f;
  for (int nl = 0; nl < Nn; ++nl) smsum += seq_mask[b * Nn + nl];
  float acc = 0.0f;
  for (int nl = 0; nl < Nn; ++nl) acc += x[(size_t)(b * Nn + nl) * EMB + col];
  pooled[i] = fmaxf(acc / smsum, 0.0f);          // h = relu(pooled)
}

__global__ void k_head1(const float* __restrict__ h, const float* __restrict__ W1,
                        const float* __restrict__ b1p, float* __restrict__ h1)
{
  int i = blockIdx.x * blockDim.x + threadIdx.x;
  if (i >= Bb * EMB) return;
  int b = i / EMB, j = i % EMB;
  float acc = b1p[j];
  for (int i2 = 0; i2 < EMB; ++i2) acc += h[b * EMB + i2] * W1[i2 * EMB + j];
  h1[i] = fmaxf(acc, 0.0f);
}

__global__ void k_out(const float* __restrict__ h1,
                      const float* __restrict__ Wm, const float* __restrict__ bmp,
                      const float* __restrict__ Wl, const float* __restrict__ blp,
                      float* __restrict__ out)
{
  int i = blockIdx.x * blockDim.x + threadIdx.x;
  if (i >= Bb * LAT) return;
  int b = i / LAT, j = i % LAT;
  float am = bmp[j], al = blp[j];
  for (int i2 = 0; i2 < EMB; ++i2) {
    float hv = h1[b * EMB + i2];
    am += hv * Wm[i2 * LAT + j];
    al += hv * Wl[i2 * LAT + j];
  }
  out[b * LAT + j] = am;
  out[Bb * LAT + b * LAT + j] = al;
}

// ============================================================================
extern "C" void kernel_launch(void* const* d_in, const int* in_sizes, int n_in,
                              void* d_out, int out_size, void* d_ws, size_t ws_size,
                              hipStream_t stream)
{
  (void)in_sizes; (void)n_in; (void)out_size; (void)ws_size;
  auto F = [&](int i) { return (const float*)d_in[i]; };
  // ---- input index map (dict insertion order) ----
  const float* coords     = F(1);
  const float* seq_mask   = F(2);
  const float* coord_mask = F(3);
  const float *neWh = F(4),  *neWs = F(5),  *neBs = F(6),  *neWv = F(7),  *neWg = F(8),  *neBg = F(9);
  const float *eeWh = F(10), *eeWs = F(11), *eeBs = F(12), *eeWv = F(13), *eeWg = F(14), *eeBg = F(15);
  const float *msgWh[NLay], *msgWs[NLay], *msgBs[NLay], *msgWv[NLay], *msgWg[NLay], *msgBg[NLay];
  const float *ffWh[NLay], *ffWs[NLay], *ffBs[NLay], *ffWv[NLay], *ffWg[NLay], *ffBg[NLay];
  const float *ln1g[NLay], *ln1b[NLay], *ln2g[NLay], *ln2b[NLay];
  for (int l = 0; l < NLay; ++l) {
    int p = 16 + 16 * l;
    msgWh[l]=F(p+0); msgWs[l]=F(p+1); msgBs[l]=F(p+2); msgWv[l]=F(p+3); msgWg[l]=F(p+4); msgBg[l]=F(p+5);
    ffWh[l]=F(p+6);  ffWs[l]=F(p+7);  ffBs[l]=F(p+8);  ffWv[l]=F(p+9);  ffWg[l]=F(p+10); ffBg[l]=F(p+11);
    ln1g[l]=F(p+12); ln1b[l]=F(p+13); ln2g[l]=F(p+14); ln2b[l]=F(p+15);
  }
  const float *Wout = F(64), *bout = F(65), *lng = F(66), *lnb = F(67);
  const float *W1 = F(68), *b1p = F(69), *Wm = F(70), *bmp = F(71), *Wl = F(72), *blp = F(73);
  float* out = (float*)d_out;

  // ---- bump allocator over d_ws ----
  char* base = (char*)d_ws; size_t off = 0;
  auto alloc = [&](size_t bytes) -> void* {
    off = (off + 255) & ~(size_t)255;
    void* r = base + off; off += bytes; return r;
  };
  auto aF = [&](size_t n) { return (float*)alloc(n * 4); };
  auto aH = [&](size_t n) { return (_Float16*)alloc(n * 2); };
  auto aI = [&](size_t n) { return (int*)alloc(n * 4); };

  // f16 transposed weights (padded [Np][Kp])
  _Float16* neWsT = aH((size_t)1024 * KNE);
  _Float16* neWvT = aH((size_t)256 * 256);
  _Float16* neWgT = aH((size_t)256 * 1024);
  _Float16* WoutT = aH((size_t)512 * 1024);
  _Float16 *msgWhT[NLay], *msgWsT[NLay], *msgWvT[NLay], *msgWgT[NLay];
  _Float16 *ffWhT[NLay], *ffWsT[NLay], *ffWvT[NLay], *ffWgT[NLay];
  for (int l = 0; l < NLay; ++l) {
    msgWhT[l] = aH((size_t)NHM * KHM);
    msgWsT[l] = aH((size_t)1024 * KMS);
    msgWvT[l] = aH((size_t)256 * KHM);
    msgWgT[l] = aH((size_t)256 * 1024);
    ffWhT[l]  = aH((size_t)256 * 256);
    ffWsT[l]  = aH((size_t)1024 * 1280);
    ffWvT[l]  = aH((size_t)256 * 256);
    ffWgT[l]  = aH((size_t)256 * 1024);
  }
  // persistent activations
  float* node_s = aF((size_t)NBn * 6);
  float* node_v = aF((size_t)NBn * 9);
  float* nf     = aF(NBn);
  int*   knn_i  = aI((size_t)NBn * Kk);
  float* w_e    = aF(NE);
  float* deg    = aF(NBn);
  _Float16* edge_s_h = aH((size_t)NE * ESz);
  float* edge_v = aF((size_t)NE * 3);
  float* s      = aF((size_t)NBn * Ssz);
  _Float16* s_h = aH((size_t)NBn * Ssz);
  float* Vv     = aF((size_t)3 * NBV);
  _Float16* Vv_h= aH((size_t)3 * NBV);
  float* agg_s  = aF((size_t)NBn * Ssz);
  float* agg_v  = aF((size_t)3 * NBV);
  // node-embed scratch
  _Float16* ne_scat = aH((size_t)NBn * KNE);
  float* neVh_f = aF((size_t)3 * NBn * 256);
  _Float16* neVh_h = aH((size_t)3 * NBn * 256);
  float* ne_so  = aF((size_t)NBn * Ssz);
  _Float16* ne_so_h = aH((size_t)NBn * Ssz);
  float* ne_gate = aF((size_t)NBn * Vsz);
  float* ne_Vo  = aF((size_t)3 * NBn * 256);
  // message-chunk scratch (L2-resident)
  _Float16* mscat = aH((size_t)MCn * KMS);
  _Float16* mVcat = aH((size_t)3 * MCn * KHM);
  float* mVh     = aF((size_t)3 * MCn * NHM);
  _Float16* mVh_h = aH((size_t)3 * MCn * NHM);
  float* mso     = aF((size_t)MCn * Ssz);
  _Float16* mso_h = aH((size_t)MCn * Ssz);
  float* mgate   = aF((size_t)MCn * Vsz);
  float* mVo     = aF((size_t)3 * MCn * Vsz);
  // feed-forward scratch
  _Float16* fscat = aH((size_t)NBn * 1280);
  float* fVh     = aF((size_t)3 * NBV);
  _Float16* fVh_h = aH((size_t)3 * NBV);
  float* fso     = aF((size_t)NBn * Ssz);
  _Float16* fso_h = aH((size_t)NBn * Ssz);
  float* fgate   = aF((size_t)NBn * Vsz);
  float* fVo     = aF((size_t)3 * NBV);
  // head scratch
  float* x_f    = aF((size_t)NBn * EMB);
  float* pooled = aF((size_t)Bb * EMB);
  float* h1     = aF((size_t)Bb * EMB);

  auto gemm = [&](const _Float16* A, const _Float16* Bt, const float* bias,
                  float* C, _Float16* Ch, int M, int N, int Kd, int lda, int ldc, int act) {
    dim3 grid(N / 64, M / 64);
    gemm_wmma<<<grid, 128, 0, stream>>>(A, Bt, bias, C, Ch, Kd, lda, ldc, act);
  };
  auto wT = [&](const float* W, _Float16* Wt, int K, int N, int Kp, int Np) {
    int tot = Np * Kp;
    k_wT<<<dim3((tot + 255) / 256), 256, 0, stream>>>(W, Wt, K, N, Kp, Np);
  };
  auto f2h = [&](const float* x, _Float16* y, int n) {
    k_f2h<<<dim3((n + 255) / 256), 256, 0, stream>>>(x, y, n);
  };

  // ---- weight conversion (f32 -> transposed, padded f16) ----
  wT(neWs, neWsT, 262, 1024, KNE, 1024);
  wT(neWv, neWvT, 256, 256, 256, 256);
  wT(neWg, neWgT, 1024, 256, 1024, 256);
  wT(Wout, WoutT, 1024, 512, 1024, 512);
  for (int l = 0; l < NLay; ++l) {
    wT(msgWh[l], msgWhT[l], 513, 513, KHM, NHM);
    wT(msgWs[l], msgWsT[l], 2593, 1024, KMS, 1024);
    wT(msgWv[l], msgWvT[l], 513, 256, KHM, 256);
    wT(msgWg[l], msgWgT[l], 1024, 256, 1024, 256);
    wT(ffWh[l],  ffWhT[l],  256, 256, 256, 256);
    wT(ffWs[l],  ffWsT[l],  1280, 1024, 1280, 1024);
    wT(ffWv[l],  ffWvT[l],  256, 256, 256, 256);
    wT(ffWg[l],  ffWgT[l],  1024, 256, 1024, 256);
  }

  // ---- geometry, KNN, edge features ----
  k_geom<<<dim3((NBn + 127) / 128), 128, 0, stream>>>(coords, seq_mask, coord_mask, node_s, node_v, nf);
  k_knn<<<dim3((NBn + 63) / 64), 64, 0, stream>>>(coords, nf, knn_i);
  k_edge<<<dim3((NE + 127) / 128), 128, 0, stream>>>(coords, nf, knn_i, eeWh, eeWs, eeBs,
                                                     eeWv, eeWg, eeBg, edge_s_h, edge_v, w_e);
  k_deg<<<dim3((NBn + 255) / 256), 256, 0, stream>>>(w_e, deg);

  // ---- node embedding GVP ----
  k_ne_pre<<<dim3((NBn * 256 + 255) / 256), 256, 0, stream>>>(node_s, node_v, neWh,
                                                              neVh_f, neVh_h, ne_scat);
  gemm(ne_scat, neWsT, neBs, ne_so, ne_so_h, NBn, 1024, KNE, KNE, 1024, 0);
  gemm(ne_so_h, neWgT, neBg, ne_gate, nullptr, NBn, 256, 1024, 1024, 256, 2);
  for (int c = 0; c < 3; ++c)
    gemm(neVh_h + (size_t)c * NBn * 256, neWvT, nullptr,
         ne_Vo + (size_t)c * NBn * 256, nullptr, NBn, 256, 256, 256, 256, 0);
  k_s_init<<<dim3((NBn * Ssz + 255) / 256), 256, 0, stream>>>(ne_so, nf, s);
  k_v_init<<<dim3((3 * NBV + 255) / 256), 256, 0, stream>>>(ne_Vo, ne_gate, nf, Vv);

  // ---- message-passing layers ----
  for (int l = 0; l < NLay; ++l) {
    f2h(s, s_h, NBn * Ssz);
    for (int ci = 0; ci < NCH; ++ci) {
      int n0 = ci * (NBn / NCH);
      k_msg_scat<<<dim3((SCAT + 255) / 256, MCn), 256, 0, stream>>>(s_h, edge_s_h, knn_i, mscat, n0);
      k_msg_vcat<<<dim3((KHM + 127) / 128, MCn, 3), 128, 0, stream>>>(Vv, edge_v, knn_i, mVcat, n0);
      for (int c = 0; c < 3; ++c)
        gemm(mVcat + (size_t)c * MCn * KHM, msgWhT[l], nullptr,
             mVh + (size_t)c * MCn * NHM, mVh_h + (size_t)c * MCn * NHM,
             MCn, NHM, KHM, KHM, NHM, 0);
      k_msg_vn<<<dim3(((KMS - SCAT) + 255) / 256, MCn), 256, 0, stream>>>(mVh, mscat);
      gemm(mscat, msgWsT[l], msgBs[l], mso, mso_h, MCn, 1024, KMS, KMS, 1024, 0);
      gemm(mso_h, msgWgT[l], msgBg[l], mgate, nullptr, MCn, 256, 1024, 1024, 256, 2);
      for (int c = 0; c < 3; ++c)
        gemm(mVh_h + (size_t)c * MCn * NHM, msgWvT[l], nullptr,
             mVo + (size_t)c * MCn * Vsz, nullptr, MCn, 256, KHM, NHM, 256, 0);
      k_agg_s<<<dim3((Ssz + 255) / 256, NBn / NCH), 256, 0, stream>>>(mso, w_e, deg, agg_s, n0);
      k_agg_v<<<dim3(1, NBn / NCH, 3), 256, 0, stream>>>(mVo, mgate, w_e, deg, agg_v, n0);
    }
    k_ln<<<dim3(NBn), 256, 0, stream>>>(s, agg_s, ln1g[l], ln1b[l], nullptr, Ssz, 0);
    k_addv<<<dim3((3 * NBV + 255) / 256), 256, 0, stream>>>(Vv, agg_v);

    // feed-forward GVP
    f2h(s, s_h, NBn * Ssz);
    f2h(Vv, Vv_h, 3 * NBV);
    for (int c = 0; c < 3; ++c)
      gemm(Vv_h + (size_t)c * NBV, ffWhT[l], nullptr,
           fVh + (size_t)c * NBV, fVh_h + (size_t)c * NBV, NBn, 256, 256, 256, 256, 0);
    k_ff_scat<<<dim3((NBn * 1280 + 255) / 256), 256, 0, stream>>>(s_h, fVh, fscat);
    gemm(fscat, ffWsT[l], ffBs[l], fso, fso_h, NBn, 1024, 1280, 1280, 1024, 0);
    gemm(fso_h, ffWgT[l], ffBg[l], fgate, nullptr, NBn, 256, 1024, 1024, 256, 2);
    for (int c = 0; c < 3; ++c)
      gemm(fVh_h + (size_t)c * NBV, ffWvT[l], nullptr,
           fVo + (size_t)c * NBV, nullptr, NBn, 256, 256, 256, 256, 0);
    k_ln<<<dim3(NBn), 256, 0, stream>>>(s, fso, ln2g[l], ln2b[l], nf, Ssz, 1);
    k_v_ff<<<dim3((3 * NBV + 255) / 256), 256, 0, stream>>>(Vv, fVo, fgate, nf);
  }

  // ---- output head ----
  f2h(s, s_h, NBn * Ssz);
  gemm(s_h, WoutT, bout, x_f, nullptr, NBn, EMB, 1024, 1024, EMB, 0);
  k_ln<<<dim3(NBn), 256, 0, stream>>>(x_f, nullptr, lng, lnb, seq_mask, EMB, 0);
  k_pool<<<dim3((Bb * EMB + 255) / 256), 256, 0, stream>>>(x_f, seq_mask, pooled);
  k_head1<<<dim3((Bb * EMB + 255) / 256), 256, 0, stream>>>(pooled, W1, b1p, h1);
  k_out<<<dim3((Bb * LAT + 255) / 256), 256, 0, stream>>>(h1, Wm, bmp, Wl, blp, out);
}
